// LSTMHMM_10273561772151
// MI455X (gfx1250) — matrix operand order repaired and checked by reading, not compiled, not run on hardware
//
#include <hip/hip_runtime.h>

// ---------------------------------------------------------------------------
// LSTM LM stack for MI455X (gfx1250, wave32, WMMA bf16 16x16x32, f32 accum)
// B=64, T=128, D=512, L=9, V=32000, S=64
//
// - Input projections / head: bf16 WMMA GEMM (bandwidth-bound, direct loads)
// - Recurrence: persistent per-layer kernel; W_hh slice staged once into LDS
//   via GLOBAL_LOAD_ASYNC_TO_LDS_B128 (ASYNCcnt), fragment-ready layout;
//   per-step B fragments come from ds_load_b128, A fragments double-buffered.
// ---------------------------------------------------------------------------

typedef __bf16 bf16_t;
typedef __bf16  v16bf __attribute__((ext_vector_type(16)));
typedef float   v8f   __attribute__((ext_vector_type(8)));
typedef unsigned u32x4 __attribute__((ext_vector_type(4)));

#define BB 64      // batch
#define TT 128     // timesteps
#define DD 512     // hidden
#define G4 2048    // 4*D
#define LL 9       // layers
#define SS 64      // head size
#define MTOK (BB*TT)   // 8192 token rows

union FragU { u32x4 q[2]; v16bf v; };

// A-matrix (16x32 bf16) fragment, row-major [*, K].
// Lanes 0-15 hold M=0..15 / K in {0..15};
// lanes 16-31 hold M=0..15 / K in {16..31}; pairs packed per dword.
__device__ __forceinline__ v16bf load_frag_a(const bf16_t* __restrict__ base,
                                             int K, int m, int k0, int lane) {
  const int c = (lane < 16) ? 0 : 16;
  const bf16_t* p = base + (size_t)m * K + (k0 + c);
  FragU f;
  f.q[0] = *reinterpret_cast<const u32x4*>(p);        // K = k0+c   .. +8
  f.q[1] = *reinterpret_cast<const u32x4*>(p + 8);    // K = k0+c+8 .. +16
  return f.v;
}

// B-matrix (32x16 bf16) fragment from weights stored [N][K] (PyTorch layout).
// Lane n = lane%16 column; lanes 0-15 K=0..15, lanes 16-31 K=16..31 (pairs/dword).
__device__ __forceinline__ v16bf load_frag_b(const bf16_t* __restrict__ W,
                                             int K, int n_base, int k0, int lane) {
  const int n  = n_base + (lane & 15);
  const int kb = (lane >> 4) << 4;                    // 0 or 16
  const bf16_t* p = W + (size_t)n * K + (k0 + kb);
  FragU f;
  f.q[0] = *reinterpret_cast<const u32x4*>(p);        // K pairs 0..7
  f.q[1] = *reinterpret_cast<const u32x4*>(p + 8);    // K pairs 8..15
  return f.v;
}

__device__ __forceinline__ float sigf(float x) { return 1.0f / (1.0f + __expf(-x)); }

// ---------------------------------------------------------------------------
// Generic bf16 WMMA GEMM with f32 bias: C[M][N] = A[M][K] * W[N][K]^T + bias
// block = 256 threads (8 waves); block tile = 128(M) x 64(N); wave = 16x64.
// ---------------------------------------------------------------------------
__global__ __launch_bounds__(256) void wmma_gemm_bias(
    const bf16_t* __restrict__ A,     // [M][K] bf16
    const bf16_t* __restrict__ W,     // [N][K] bf16
    const float*  __restrict__ bias,  // [N]
    float* __restrict__ C,            // [M][N]
    int M, int N, int K)
{
  const int lane  = threadIdx.x & 31;
  const int wave  = threadIdx.x >> 5;             // 0..7
  const int m0    = blockIdx.y * 128 + wave * 16;
  const int n0    = blockIdx.x * 64;
  const int nl    = lane & 15;
  const int moff  = (lane >> 4) * 8;

  v8f acc[4];
#pragma unroll
  for (int j = 0; j < 4; ++j) {
    const float bv = bias[n0 + 16 * j + nl];
#pragma unroll
    for (int r = 0; r < 8; ++r) acc[j][r] = bv;
  }

  for (int k0 = 0; k0 < K; k0 += 32) {
    const v16bf a = load_frag_a(A, K, m0 + nl, k0, lane);
#pragma unroll
    for (int j = 0; j < 4; ++j) {
      const v16bf b = load_frag_b(W, K, n0 + 16 * j, k0, lane);
      acc[j] = __builtin_amdgcn_wmma_f32_16x16x32_bf16(
          false, a, false, b, (short)0, acc[j], false, false);
    }
  }

#pragma unroll
  for (int j = 0; j < 4; ++j)
#pragma unroll
    for (int r = 0; r < 8; ++r)
      C[(size_t)(m0 + moff + r) * N + (n0 + 16 * j + nl)] = acc[j][r];
}

// ---------------------------------------------------------------------------
// Software grid barrier (32 scan blocks, all resident).
// ---------------------------------------------------------------------------
__device__ __forceinline__ void grid_barrier(unsigned* bar, unsigned target) {
  __syncthreads();
  if (threadIdx.x == 0) {
    __hip_atomic_fetch_add(bar, 1u, __ATOMIC_ACQ_REL, __HIP_MEMORY_SCOPE_AGENT);
    while (__hip_atomic_load(bar, __ATOMIC_ACQUIRE, __HIP_MEMORY_SCOPE_AGENT) < target) {
      __builtin_amdgcn_s_sleep(1);
    }
  }
  __syncthreads();
}

// ---------------------------------------------------------------------------
// Persistent LSTM scan for one layer. Grid = 32 blocks x 256 threads.
// Block b owns hidden columns [b*16, b*16+16) and computes, per timestep,
// the recurrent GEMM for all 4 gates of those columns (16 WMMA tiles:
// 4 m-tiles x 4 gates, 2 tiles per wave), so gate fusion is block-local.
// The block's W_hh slice (64 KB) is staged ONCE into LDS (async copies) in a
// fragment-ready layout; each per-step B fragment is then two ds_load_b128.
// h state is ping-ponged so only 1 grid barrier/step is needed.
// ---------------------------------------------------------------------------
__global__ __launch_bounds__(256) void lstm_scan(
    const bf16_t* __restrict__ Whh,     // [2048][512] bf16
    const float*  __restrict__ xg,      // [64][128][2048] f32 (bias included)
    float* __restrict__ c_state,        // [64][512] f32 (zero-init)
    bf16_t* __restrict__ hping,         // [2][64][512] bf16 (zero-init)
    float* __restrict__ h_out,          // [64][128][512] f32
    bf16_t* __restrict__ h_out_bf,      // [64][128][512] bf16 (next layer A)
    unsigned* __restrict__ bar)
{
  // Fragment-ready W_hh slice: [gate][k-chunk][frag-half][lane] of 16B each.
  // Lane-major 16B units -> each half-wave ds_load_b128 is bank-conflict free.
  __shared__ u32x4 wfrag[4][16][2][32];   // 64 KB
  __shared__ float gates[BB][4][16];      // 16 KB

  const int lane   = threadIdx.x & 31;
  const int wave   = threadIdx.x >> 5;  // 0..7
  const int d0     = blockIdx.x * 16;   // this block's hidden columns
  const int m0     = (wave & 3) * 16;   // rows for this wave's tiles
  const int g0     = (wave >> 2) * 2;   // gates {g0, g0+1}
  const int nl     = lane & 15;
  const int moff   = (lane >> 4) * 8;

  // ---- Stage W_hh block slice into LDS with async global->LDS copies ----
  // 4096 x 16B transfers; each thread contributes one 16B unit per pass.
#pragma unroll
  for (int i = 0; i < 16; ++i) {
    const int f  = threadIdx.x + 256 * i;    // index into [g][c][h][lane]
    const int li = f & 31;                   // == lane (256 % 32 == 0)
    const int hh = (f >> 5) & 1;
    const int cc = (f >> 6) & 15;
    const int gg = (f >> 10) & 3;
    const int n  = gg * DD + d0 + (li & 15);
    const int k  = cc * 32 + ((li >> 4) << 4) + hh * 8;
    const bf16_t* src = Whh + (size_t)n * DD + k;
    const unsigned dst = (unsigned)(size_t)&wfrag[gg][cc][hh][li];
    asm volatile("global_load_async_to_lds_b128 %0, %1, off"
                 :: "v"(dst), "v"(src) : "memory");
  }
  asm volatile("s_wait_asynccnt 0x0" ::: "memory");
  __syncthreads();

  for (int t = 0; t < TT; ++t) {
    const bf16_t* H = hping + (size_t)(t & 1) * BB * DD;

    // Accumulators seeded with precomputed input projection xg[:, t, :]
    v8f acc0, acc1;
    const int na = g0 * DD + d0 + nl;
    const int nb = (g0 + 1) * DD + d0 + nl;
#pragma unroll
    for (int r = 0; r < 8; ++r) {
      const size_t row = (size_t)(m0 + moff + r) * TT + t;
      acc0[r] = xg[row * G4 + na];
      acc1[r] = xg[row * G4 + nb];
    }
    if (t + 1 < TT) {  // warm L2/WGP$ for next step's seed loads
      __builtin_prefetch(&xg[((size_t)(m0 + moff) * TT + t + 1) * G4 + na], 0, 1);
      __builtin_prefetch(&xg[((size_t)(m0 + moff) * TT + t + 1) * G4 + nb], 0, 1);
    }

    // Recurrent GEMM: A (h state) double-buffered from global, B from LDS.
    v16bf a_cur = load_frag_a(H, DD, m0 + nl, 0, lane);
#pragma unroll 4
    for (int c = 0; c < 16; ++c) {
      const v16bf a_nxt = load_frag_a(H, DD, m0 + nl, ((c + 1) & 15) * 32, lane);
      FragU fb0, fb1;
      fb0.q[0] = wfrag[g0][c][0][lane];
      fb0.q[1] = wfrag[g0][c][1][lane];
      fb1.q[0] = wfrag[g0 + 1][c][0][lane];
      fb1.q[1] = wfrag[g0 + 1][c][1][lane];
      acc0 = __builtin_amdgcn_wmma_f32_16x16x32_bf16(
          false, a_cur, false, fb0.v, (short)0, acc0, false, false);
      acc1 = __builtin_amdgcn_wmma_f32_16x16x32_bf16(
          false, a_cur, false, fb1.v, (short)0, acc1, false, false);
      a_cur = a_nxt;
    }

#pragma unroll
    for (int r = 0; r < 8; ++r) {
      gates[m0 + moff + r][g0][nl]     = acc0[r];
      gates[m0 + moff + r][g0 + 1][nl] = acc1[r];
    }
    __syncthreads();

    // Fused LSTM cell update: 64 rows x 16 cols = 1024 elems / 256 threads
#pragma unroll
    for (int j = 0; j < 4; ++j) {
      const int idx = threadIdx.x + 256 * j;
      const int b   = idx >> 4;
      const int dl  = idx & 15;
      const int d   = d0 + dl;
      const float ig = gates[b][0][dl];
      const float fg = gates[b][1][dl];
      const float gg = gates[b][2][dl];
      const float og = gates[b][3][dl];
      const float c_old = c_state[b * DD + d];
      const float cn = sigf(fg) * c_old + sigf(ig) * tanhf(gg);
      const float hn = sigf(og) * tanhf(cn);
      c_state[b * DD + d] = cn;
      const size_t oidx = ((size_t)b * TT + t) * DD + d;
      h_out[oidx]    = hn;
      h_out_bf[oidx] = (bf16_t)hn;
      hping[(size_t)((t + 1) & 1) * BB * DD + b * DD + d] = (bf16_t)hn;
    }
    grid_barrier(bar, 32u * (unsigned)(t + 1));
  }
}

// ---------------------------------------------------------------------------
// Support kernels
// ---------------------------------------------------------------------------
__global__ void f32_to_bf16_kernel(const float* __restrict__ src,
                                   bf16_t* __restrict__ dst, int n) {
  const int i = blockIdx.x * blockDim.x + threadIdx.x;
  if (i < n) dst[i] = (bf16_t)src[i];
}

__global__ __launch_bounds__(256) void embed_kernel(
    const int* __restrict__ x, const float* __restrict__ emb,
    bf16_t* __restrict__ out) {
  const int m = blockIdx.x;                 // token row 0..8191
  const int tok = x[m];
  const float* e = emb + (size_t)tok * DD;
  out[(size_t)m * DD + threadIdx.x]       = (bf16_t)e[threadIdx.x];
  out[(size_t)m * DD + threadIdx.x + 256] = (bf16_t)e[threadIdx.x + 256];
}

__global__ void init_state_kernel(float* __restrict__ c_state,
                                  bf16_t* __restrict__ hping,
                                  unsigned* __restrict__ bar) {
  const int i = blockIdx.x * blockDim.x + threadIdx.x;
  if (i == 0) *bar = 0u;
  if (i < BB * DD) c_state[i] = 0.0f;
  if (i < 2 * BB * DD) hping[i] = (bf16_t)0.0f;
}

__global__ __launch_bounds__(256) void layernorm_kernel(
    const float* __restrict__ h,      // [8192][512]
    const float* __restrict__ gamma, const float* __restrict__ beta,
    float* __restrict__ hn_out,       // [8192][512] (into d_out)
    bf16_t* __restrict__ hn_bf)       // [8192][512] bf16 for head GEMM
{
  __shared__ float red0[256];
  __shared__ float red1[256];
  const int m = blockIdx.x;
  const float* row = h + (size_t)m * DD;
  const float x0 = row[threadIdx.x];
  const float x1 = row[threadIdx.x + 256];
  red0[threadIdx.x] = x0 + x1;
  red1[threadIdx.x] = x0 * x0 + x1 * x1;
  __syncthreads();
  for (int off = 128; off > 0; off >>= 1) {
    if (threadIdx.x < off) {
      red0[threadIdx.x] += red0[threadIdx.x + off];
      red1[threadIdx.x] += red1[threadIdx.x + off];
    }
    __syncthreads();
  }
  const float mean = red0[0] * (1.0f / DD);
  const float var  = red1[0] * (1.0f / DD) - mean * mean;
  const float rs   = __frsqrt_rn(var + 1e-5f);
  const float y0 = (x0 - mean) * rs * gamma[threadIdx.x]       + beta[threadIdx.x];
  const float y1 = (x1 - mean) * rs * gamma[threadIdx.x + 256] + beta[threadIdx.x + 256];
  hn_out[(size_t)m * DD + threadIdx.x]       = y0;
  hn_out[(size_t)m * DD + threadIdx.x + 256] = y1;
  hn_bf[(size_t)m * DD + threadIdx.x]        = (bf16_t)y0;
  hn_bf[(size_t)m * DD + threadIdx.x + 256]  = (bf16_t)y1;
}

// ---------------------------------------------------------------------------
// Host orchestration
// ---------------------------------------------------------------------------
extern "C" void kernel_launch(void* const* d_in, const int* in_sizes, int n_in,
                              void* d_out, int out_size, void* d_ws, size_t ws_size,
                              hipStream_t stream) {
  (void)in_sizes; (void)n_in; (void)out_size; (void)ws_size;

  const int*   x      = (const int*)  d_in[0];   // [B,T]
  const float* emb    = (const float*)d_in[1];   // [V,D]
  const float* W_ih   = (const float*)d_in[2];   // [L,4D,D]
  const float* W_hh   = (const float*)d_in[3];   // [L,4D,D]
  const float* bias   = (const float*)d_in[4];   // [L,4D]
  const float* gamma  = (const float*)d_in[5];   // [D]
  const float* beta   = (const float*)d_in[6];   // [D]
  const float* head_W = (const float*)d_in[7];   // [S,D]
  const float* head_b = (const float*)d_in[8];   // [S]

  // Workspace carve-up (all chunk sizes are multiples of 256 B)
  char* ws = (char*)d_ws;
  size_t off = 0;
  auto take = [&](size_t bytes) { char* p = ws + off; off += bytes; return p; };
  const size_t nWl = (size_t)G4 * DD;                      // per-layer weight elems
  bf16_t* Wih_bf  = (bf16_t*)take(LL * nWl * sizeof(bf16_t));
  bf16_t* Whh_bf  = (bf16_t*)take(LL * nWl * sizeof(bf16_t));
  bf16_t* headW_bf= (bf16_t*)take((size_t)SS * DD * sizeof(bf16_t));
  bf16_t* hA      = (bf16_t*)take((size_t)MTOK * DD * sizeof(bf16_t));
  bf16_t* hB      = (bf16_t*)take((size_t)MTOK * DD * sizeof(bf16_t));
  float*  xg      = (float*) take((size_t)MTOK * G4 * sizeof(float));
  float*  h_f32   = (float*) take((size_t)MTOK * DD * sizeof(float));
  bf16_t* hn_bf   = (bf16_t*)take((size_t)MTOK * DD * sizeof(bf16_t));
  float*  c_state = (float*) take((size_t)BB * DD * sizeof(float));
  bf16_t* hping   = (bf16_t*)take((size_t)2 * BB * DD * sizeof(bf16_t));
  unsigned* bar   = (unsigned*)take(256);

  // Weight conversion to bf16 (kept in [N][K] layout for B-fragment loads)
  const int nW = LL * (int)nWl;
  f32_to_bf16_kernel<<<(nW + 255) / 256, 256, 0, stream>>>(W_ih, Wih_bf, nW);
  f32_to_bf16_kernel<<<(nW + 255) / 256, 256, 0, stream>>>(W_hh, Whh_bf, nW);
  f32_to_bf16_kernel<<<(SS * DD + 255) / 256, 256, 0, stream>>>(head_W, headW_bf, SS * DD);

  // Embedding -> bf16 activations
  embed_kernel<<<MTOK, 256, 0, stream>>>(x, emb, hA);

  bf16_t* Ain = hA;
  bf16_t* Anext = hB;
  for (int l = 0; l < LL; ++l) {
    init_state_kernel<<<(2 * BB * DD + 255) / 256, 256, 0, stream>>>(c_state, hping, bar);
    // xg = h @ W_ih^T + b : M=8192, N=2048, K=512
    wmma_gemm_bias<<<dim3(G4 / 64, MTOK / 128), 256, 0, stream>>>(
        Ain, Wih_bf + (size_t)l * nWl, bias + (size_t)l * G4, xg, MTOK, G4, DD);
    // persistent recurrent scan over T=128 steps
    lstm_scan<<<32, 256, 0, stream>>>(
        Whh_bf + (size_t)l * nWl, xg, c_state, hping, h_f32, Anext, bar);
    bf16_t* tmp = Ain; Ain = Anext; Anext = tmp;
  }

  float* out    = (float*)d_out;
  float* logits = out;                       // [8192][64]
  float* hn_out = out + (size_t)MTOK * SS;   // [8192][512]

  layernorm_kernel<<<MTOK, 256, 0, stream>>>(h_f32, gamma, beta, hn_out, hn_bf);
  // logits = hn @ head_W^T + head_b : M=8192, N=64, K=512
  wmma_gemm_bias<<<dim3(SS / 64, MTOK / 128), 256, 0, stream>>>(
      hn_bf, headW_bf, head_b, logits, MTOK, SS, DD);
}
